// SpatialAggregator_79439715107378
// MI455X (gfx1250) — compile-verified, matrix operand order checked by reading, not run on hardware
//
#include <hip/hip_runtime.h>

typedef _Float16 v16h __attribute__((ext_vector_type(16)));
typedef _Float16 v8h  __attribute__((ext_vector_type(8)));
typedef float    v8f  __attribute__((ext_vector_type(8)));

#define WMMA(a, b, c) __builtin_amdgcn_wmma_f32_16x16x32_f16( \
    false, (a), false, (b), (short)0, (c), false, false)

// Async L2->LDS copy of 16 bytes per lane (CDNA5 GLOBAL_LOAD_ASYNC_TO_LDS_B128).
__device__ inline void async_cp16(unsigned ldsOff, const void* gaddr) {
  unsigned long long ga = (unsigned long long)(uintptr_t)gaddr;
  asm volatile("global_load_async_to_lds_b128 %0, %1, off"
               :: "v"(ldsOff), "v"(ga) : "memory");
}
__device__ inline void async_wait0() {
  asm volatile("s_wait_asynccnt 0x0" ::: "memory");
}
__device__ inline void async_wait8() {  // allow the 8 just-issued copies in flight
  asm volatile("s_wait_asynccnt 0x8" ::: "memory");
}
__device__ inline unsigned lds_off(const void* p) {
  return (unsigned)(uintptr_t)p;  // shared aperture: addr[31:0] = LDS offset
}

// Load one 16x32 f16 fragment (A or B operand) from an LDS tile stored
// [row][ld] row-major ("row" = M for A / N for B, columns = K).
// Lane layout per CDNA5 ISA 7.12.2.
__device__ inline v16h frag_ld(const _Float16* t, int row0, int ld, int kofs) {
  int lane = threadIdx.x & 31;
  int hi = lane >> 4;
  const _Float16* p = t + (row0 + (lane & 15)) * ld + kofs + hi * 8;
  v8h a = *(const v8h*)p;
  v8h b = *(const v8h*)(p + 16);
  return __builtin_shufflevector(a, b, 0, 1, 2, 3, 4, 5, 6, 7,
                                 8, 9, 10, 11, 12, 13, 14, 15);
}

// ---------------- pos-embed add + NCHW -> NHWC (f32 + f16 copy) ----------------
__global__ void prep_kernel(const float* __restrict__ x,
                            const float* __restrict__ ph,
                            const float* __restrict__ pw,
                            float* __restrict__ y, _Float16* __restrict__ yh) {
  int idx = blockIdx.x * blockDim.x + threadIdx.x;  // B*H*W*C order
  int c = idx & 511;
  int w = (idx >> 9) & 63;
  int h = (idx >> 15) & 63;
  int b = idx >> 21;
  float v = x[((b * 512 + c) * 64 + h) * 64 + w] + ph[c * 64 + h] + pw[c * 64 + w];
  y[idx] = v;
  yh[idx] = (_Float16)v;
}

// ---------------- weight convert f32 -> f16 WITH transpose ----------------
// in: (8, 512, N) f32 row-major; out: (8, N, 512) f16 ([n][k] per matrix)
__global__ void cvt_t_kernel(const float* __restrict__ in,
                             _Float16* __restrict__ out, int N, int lgN) {
  int idx = blockIdx.x * blockDim.x + threadIdx.x;  // over 8*N*512
  int k = idx & 511;
  int nm = idx >> 9;
  int n = nm & (N - 1);
  int mat = nm >> lgN;
  out[idx] = (_Float16)in[(size_t)(mat * 512 + k) * N + n];
}

// ---------------- GEMM: C_f16[M,N] = A_f16[M,512] * Bt_f16[N,512] ----------------
// 128x128 block tile, BK=32, 4 waves each own 64x64; double-buffered async LDS.
__global__ __launch_bounds__(128) void gemm_qkv(const _Float16* __restrict__ A,
                                                const _Float16* __restrict__ Bt,
                                                _Float16* __restrict__ C, int N) {
  __shared__ alignas(16) _Float16 As[2][128 * 32];  // [m][k]
  __shared__ alignas(16) _Float16 Bs[2][128 * 32];  // [n][k]
  int tid = threadIdx.x;
  int m0 = blockIdx.x * 128;
  int n0 = blockIdx.y * 128;
  int wv = tid >> 5;
  int wm = (wv >> 1) * 64, wn = (wv & 1) * 64;
  unsigned aB[2] = {lds_off(As[0]), lds_off(As[1])};
  unsigned bB[2] = {lds_off(Bs[0]), lds_off(Bs[1])};
  v8f acc[4][4] = {};
#pragma unroll
  for (int t = 0; t < 4; ++t) {  // preamble: tile for kt=0 into buffer 0
    int ci = tid + t * 128;
    int r = ci >> 2, kk = (ci & 3) * 8;
    unsigned so = (unsigned)(r * 32 + kk) * 2;
    async_cp16(aB[0] + so, &A[(size_t)(m0 + r) * 512 + kk]);
    async_cp16(bB[0] + so, &Bt[(size_t)(n0 + r) * 512 + kk]);
  }
  for (int it = 0; it < 16; ++it) {
    int buf = it & 1;
    if (it < 15) {
      int kt = (it + 1) * 32;
#pragma unroll
      for (int t = 0; t < 4; ++t) {
        int ci = tid + t * 128;
        int r = ci >> 2, kk = (ci & 3) * 8;
        unsigned so = (unsigned)(r * 32 + kk) * 2;
        async_cp16(aB[1 - buf] + so, &A[(size_t)(m0 + r) * 512 + kt + kk]);
        async_cp16(bB[1 - buf] + so, &Bt[(size_t)(n0 + r) * 512 + kt + kk]);
      }
      async_wait8();  // current tile landed; next tile still in flight
    } else {
      async_wait0();
    }
    __syncthreads();
    v16h af[4], bf[4];
#pragma unroll
    for (int i = 0; i < 4; ++i) af[i] = frag_ld(As[buf], wm + i * 16, 32, 0);
#pragma unroll
    for (int j = 0; j < 4; ++j) bf[j] = frag_ld(Bs[buf], wn + j * 16, 32, 0);
#pragma unroll
    for (int i = 0; i < 4; ++i)
#pragma unroll
      for (int j = 0; j < 4; ++j) acc[i][j] = WMMA(af[i], bf[j], acc[i][j]);
    __syncthreads();
  }
  int lane = tid & 31;
  int rofs = (lane >> 4) * 8, cofs = lane & 15;
#pragma unroll
  for (int i = 0; i < 4; ++i)
#pragma unroll
    for (int j = 0; j < 4; ++j)
#pragma unroll
      for (int r = 0; r < 8; ++r) {
        int row = m0 + wm + i * 16 + rofs + r;
        int col = n0 + wn + j * 16 + cofs;
        C[(size_t)row * N + col] = (_Float16)acc[i][j][r];
      }
}

// -- GEMM: out_f32[M,512] (=|+=) A_f16*WoT_f16 + bias; accum pass also emits f16 --
__global__ __launch_bounds__(128) void gemm_out(const _Float16* __restrict__ A,
                                                const _Float16* __restrict__ Bt,
                                                const float* __restrict__ bias,
                                                float* __restrict__ Cout,
                                                _Float16* __restrict__ Ch,
                                                int accum) {
  const int N = 512;
  __shared__ alignas(16) _Float16 As[2][128 * 32];
  __shared__ alignas(16) _Float16 Bs[2][128 * 32];
  int tid = threadIdx.x;
  int m0 = blockIdx.x * 128;
  int n0 = blockIdx.y * 128;
  int wv = tid >> 5;
  int wm = (wv >> 1) * 64, wn = (wv & 1) * 64;
  unsigned aB[2] = {lds_off(As[0]), lds_off(As[1])};
  unsigned bB[2] = {lds_off(Bs[0]), lds_off(Bs[1])};
  v8f acc[4][4] = {};
#pragma unroll
  for (int t = 0; t < 4; ++t) {
    int ci = tid + t * 128;
    int r = ci >> 2, kk = (ci & 3) * 8;
    unsigned so = (unsigned)(r * 32 + kk) * 2;
    async_cp16(aB[0] + so, &A[(size_t)(m0 + r) * 512 + kk]);
    async_cp16(bB[0] + so, &Bt[(size_t)(n0 + r) * 512 + kk]);
  }
  for (int it = 0; it < 16; ++it) {
    int buf = it & 1;
    if (it < 15) {
      int kt = (it + 1) * 32;
#pragma unroll
      for (int t = 0; t < 4; ++t) {
        int ci = tid + t * 128;
        int r = ci >> 2, kk = (ci & 3) * 8;
        unsigned so = (unsigned)(r * 32 + kk) * 2;
        async_cp16(aB[1 - buf] + so, &A[(size_t)(m0 + r) * 512 + kt + kk]);
        async_cp16(bB[1 - buf] + so, &Bt[(size_t)(n0 + r) * 512 + kt + kk]);
      }
      async_wait8();
    } else {
      async_wait0();
    }
    __syncthreads();
    v16h af[4], bf[4];
#pragma unroll
    for (int i = 0; i < 4; ++i) af[i] = frag_ld(As[buf], wm + i * 16, 32, 0);
#pragma unroll
    for (int j = 0; j < 4; ++j) bf[j] = frag_ld(Bs[buf], wn + j * 16, 32, 0);
#pragma unroll
    for (int i = 0; i < 4; ++i)
#pragma unroll
      for (int j = 0; j < 4; ++j) acc[i][j] = WMMA(af[i], bf[j], acc[i][j]);
    __syncthreads();
  }
  int lane = tid & 31;
  int rofs = (lane >> 4) * 8, cofs = lane & 15;
#pragma unroll
  for (int i = 0; i < 4; ++i)
#pragma unroll
    for (int j = 0; j < 4; ++j)
#pragma unroll
      for (int r = 0; r < 8; ++r) {
        int row = m0 + wm + i * 16 + rofs + r;
        int col = n0 + wn + j * 16 + cofs;
        size_t idx = (size_t)row * N + col;
        float v = acc[i][j][r] + bias[col];
        if (accum) {
          float s = Cout[idx] + v;     // oh + ow
          Cout[idx] = s;
          Ch[idx] = (_Float16)s;       // f16 activation copy for next layer's GEMMs
        } else {
          Cout[idx] = v;
        }
      }
}

// ---------------- per-(sequence, head) attention, L=64, e=64 ----------------
// grid.x = 512 seqs * 8 heads; 4 waves; each wave owns a 16-row strip.
__global__ __launch_bounds__(128) void attn_kernel(const _Float16* __restrict__ q,
                                                   const _Float16* __restrict__ kv,
                                                   _Float16* __restrict__ o,
                                                   int posMul, int rowStride) {
  __shared__ alignas(16) _Float16 Qs[64 * 64];  // [i][e]
  __shared__ alignas(16) _Float16 Ks[64 * 64];  // [j][e]  (B operand for S=QK^T)
  __shared__ alignas(16) _Float16 Vt[64 * 64];  // [e][j]  (B operand for O=PV)
  __shared__ alignas(16) _Float16 Ps[64 * 64];  // [i][j]
  int head = blockIdx.x & 7;
  int seq = blockIdx.x >> 3;
  int b = seq >> 6, s = seq & 63;
  int baseRow = b * 4096 + s * posMul;
  int tid = threadIdx.x;
  int qcol = head * 64;
  unsigned qsBase = lds_off(Qs), ksBase = lds_off(Ks);

  for (int ci = tid; ci < 512; ci += 128) {
    int p = ci >> 3, ec = (ci & 7) * 8;
    size_t gr = (size_t)(baseRow + p * rowStride) * 512;
    unsigned so = (unsigned)(p * 64 + ec) * 2;
    async_cp16(qsBase + so, &q[gr + qcol + ec]);
    async_cp16(ksBase + so, &kv[gr * 2 + qcol + ec]);
    v8h vv = *(const v8h*)&kv[gr * 2 + 512 + qcol + ec];
#pragma unroll
    for (int u = 0; u < 8; ++u) Vt[(ec + u) * 64 + p] = vv[u];
  }
  async_wait0();
  __syncthreads();

  int wv = tid >> 5, lane = tid & 31;
  int m0 = wv * 16;
  int rofs = (lane >> 4) * 8, cofs = lane & 15;

  v8f sacc[4] = {};
  for (int ks = 0; ks < 64; ks += 32) {
    v16h af = frag_ld(Qs, m0, 64, ks);
#pragma unroll
    for (int j = 0; j < 4; ++j) {
      v16h bf = frag_ld(Ks, j * 16, 64, ks);
      sacc[j] = WMMA(af, bf, sacc[j]);
    }
  }
  // softmax: element r of each acc VGPR is row m0+rofs+r; its 16 columns for
  // tile j live across this half-wave's 16 lanes -> shfl_xor(1,2,4,8) reduces.
#pragma unroll
  for (int r = 0; r < 8; ++r) {
    float mx = -3.0e38f;
#pragma unroll
    for (int j = 0; j < 4; ++j) {
      sacc[j][r] *= 0.125f;  // e^-0.5, e=64
      mx = fmaxf(mx, sacc[j][r]);
    }
    for (int off = 1; off < 16; off <<= 1) mx = fmaxf(mx, __shfl_xor(mx, off, 32));
    float sum = 0.0f;
#pragma unroll
    for (int j = 0; j < 4; ++j) {
      float e = __expf(sacc[j][r] - mx);
      sacc[j][r] = e;
      sum += e;
    }
    for (int off = 1; off < 16; off <<= 1) sum += __shfl_xor(sum, off, 32);
    float rs = 1.0f / sum;
    int m = m0 + rofs + r;
#pragma unroll
    for (int j = 0; j < 4; ++j)
      Ps[m * 64 + j * 16 + cofs] = (_Float16)(sacc[j][r] * rs);
  }
  __syncthreads();

  v8f oacc[4] = {};
  for (int ks = 0; ks < 64; ks += 32) {
    v16h af = frag_ld(Ps, m0, 64, ks);
#pragma unroll
    for (int j = 0; j < 4; ++j) {
      v16h bf = frag_ld(Vt, j * 16, 64, ks);
      oacc[j] = WMMA(af, bf, oacc[j]);
    }
  }
#pragma unroll
  for (int j = 0; j < 4; ++j)
#pragma unroll
    for (int r = 0; r < 8; ++r) {
      int m = m0 + rofs + r;
      size_t gr = (size_t)(baseRow + m * rowStride) * 512;
      o[gr + qcol + j * 16 + cofs] = (_Float16)oacc[j][r];
    }
}

// ---------------- NHWC f32 -> NCHW output ----------------
__global__ void fin_kernel(const float* __restrict__ y, float* __restrict__ out) {
  int idx = blockIdx.x * blockDim.x + threadIdx.x;  // B*C*H*W order
  int w = idx & 63;
  int h = (idx >> 6) & 63;
  int c = (idx >> 12) & 511;
  int b = idx >> 21;
  out[idx] = y[((b * 64 + h) * 64 + w) * 512 + c];
}

extern "C" void kernel_launch(void* const* d_in, const int* in_sizes, int n_in,
                              void* d_out, int out_size, void* d_ws, size_t ws_size,
                              hipStream_t stream) {
  const float* x = (const float*)d_in[0];
  const float* ph = (const float*)d_in[1];
  const float* pw = (const float*)d_in[2];
  const float* wq = (const float*)d_in[3];
  const float* wkv = (const float*)d_in[4];
  const float* wo = (const float*)d_in[5];
  const float* wob = (const float*)d_in[6];

  const size_t M = 32768;  // B*H*W
  char* ws = (char*)d_ws;
  size_t off = 0;
  float* yA = (float*)(ws + off); off += M * 512 * 4;
  float* yB = (float*)(ws + off); off += M * 512 * 4;
  _Float16* yh = (_Float16*)(ws + off); off += M * 512 * 2;   // f16 activation copy
  _Float16* qbuf = (_Float16*)(ws + off); off += M * 512 * 2;
  _Float16* kvbuf = (_Float16*)(ws + off); off += M * 1024 * 2;
  _Float16* wqT = (_Float16*)(ws + off); off += (size_t)8 * 512 * 512 * 2;
  _Float16* wkvT = (_Float16*)(ws + off); off += (size_t)8 * 512 * 1024 * 2;
  _Float16* woT = (_Float16*)(ws + off); off += (size_t)8 * 512 * 512 * 2;
  _Float16* aobuf = qbuf;  // alias: attn block reads exactly the region it writes

  prep_kernel<<<16777216 / 256, 256, 0, stream>>>(x, ph, pw, yA, yh);
  cvt_t_kernel<<<2097152 / 256, 256, 0, stream>>>(wq, wqT, 512, 9);
  cvt_t_kernel<<<4194304 / 256, 256, 0, stream>>>(wkv, wkvT, 1024, 10);
  cvt_t_kernel<<<2097152 / 256, 256, 0, stream>>>(wo, woT, 512, 9);

  for (int l = 0; l < 4; ++l) {
    float* ynext = (l & 1) ? yA : yB;
    for (int axis = 0; axis < 2; ++axis) {
      int la = l * 2 + axis;
      gemm_qkv<<<dim3(256, 4), 128, 0, stream>>>(
          yh, wqT + (size_t)la * 512 * 512, qbuf, 512);
      gemm_qkv<<<dim3(256, 8), 128, 0, stream>>>(
          yh, wkvT + (size_t)la * 512 * 1024, kvbuf, 1024);
      int posMul = axis ? 64 : 1;     // height: base = b*4096 + w ; width: b*4096 + h*64
      int rowStride = axis ? 1 : 64;  // height attends with row stride W
      attn_kernel<<<4096, 128, 0, stream>>>(qbuf, kvbuf, aobuf, posMul, rowStride);
      // axis 1 accumulates oh+ow and refreshes the f16 activation copy
      gemm_out<<<dim3(256, 4), 128, 0, stream>>>(
          aobuf, woT + (size_t)la * 512 * 512, wob + (size_t)la * 512,
          ynext, yh, axis);
    }
  }
  float* yfin = (4 & 1) ? yB : yA;  // layer 3 wrote yA
  fin_kernel<<<16777216 / 256, 256, 0, stream>>>(yfin, (float*)d_out);
}
